// Gate_78228534329540
// MI455X (gfx1250) — matrix operand order repaired and checked by reading, not compiled, not run on hardware
//
#include <hip/hip_runtime.h>

typedef __bf16 bf16;
typedef bf16  bf16x4  __attribute__((ext_vector_type(4)));
typedef bf16  bf16x8  __attribute__((ext_vector_type(8)));
typedef bf16  bf16x16 __attribute__((ext_vector_type(16)));
typedef float floatx4 __attribute__((ext_vector_type(4)));
typedef float v8f     __attribute__((ext_vector_type(8)));

static constexpr int   D_DIM = 4096;
static constexpr int   E_DIM = 256;
static constexpr int   TOPK  = 6;
static constexpr float ROUTE_SCALE = 1.5f;
static constexpr int   CHUNK = 256;   // K-chunk staged in LDS
static constexpr int   ROWS  = 16;    // rows of x per block

__device__ __forceinline__ void split_bf16(float v, bf16& h, bf16& l) {
    h = (bf16)v;                 // RNE high part
    l = (bf16)(v - (float)h);    // residual
}

// ---- prep: split weight fp32 -> bf16 hi/lo planes in workspace (4 MB) ----
__global__ __launch_bounds__(256)
void split_w_kernel(const float* __restrict__ w,
                    bf16* __restrict__ whi, bf16* __restrict__ wlo)
{
    const size_t i = ((size_t)blockIdx.x * 256 + threadIdx.x) * 4;
    floatx4 v = *(const floatx4*)(w + i);
    bf16x4 h, l;
    #pragma unroll
    for (int j = 0; j < 4; ++j) { bf16 hh, ll; split_bf16(v[j], hh, ll); h[j]=hh; l[j]=ll; }
    *(bf16x4*)(whi + i) = h;
    *(bf16x4*)(wlo + i) = l;
}

template <bool PRECONV>
__global__ __launch_bounds__(512)
void gate_topk_kernel(const float* __restrict__ x,
                      const float* __restrict__ w,
                      const float* __restrict__ bias,
                      const bf16* __restrict__ whi,
                      const bf16* __restrict__ wlo,
                      float* __restrict__ out_w,
                      float* __restrict__ out_i)
{
    __shared__ bf16  s_hi[ROWS * CHUNK];     // 8 KB
    __shared__ bf16  s_lo[ROWS * CHUNK];     // 8 KB
    __shared__ float s_orig[ROWS * E_DIM];   // 16 KB
    __shared__ float s_bias[E_DIM];          // 1 KB

    const int tid  = threadIdx.x;
    const int lane = tid & 31;
    const int wave = tid >> 5;               // 0..15 (16 waves of 32)
    const int r0   = blockIdx.x * ROWS;      // global row base
    const int e0   = wave * 16;              // expert tile base for this wave

    if (tid < E_DIM) s_bias[tid] = bias[tid];

    // ---- GEMM: scores[16 rows][16 experts] per wave, split-bf16 x3 WMMA ----
    v8f acc = {0.f,0.f,0.f,0.f,0.f,0.f,0.f,0.f};

    const int m    = lane & 15;              // A row / B column within tile
    const int hi16 = lane >> 4;
    const int aoff = hi16 ? 16 : 0;          // A lane K-segment select
    const int bko  = hi16 ? 16 : 0;          // B lane K-base select
    const size_t wbase = (size_t)(e0 + m) * D_DIM;
    const float* wrow  = w + wbase;          // fallback path
    const bf16*  whrow = PRECONV ? whi + wbase : nullptr;
    const bf16*  wlrow = PRECONV ? wlo + wbase : nullptr;

    for (int k0 = 0; k0 < D_DIM; k0 += CHUNK) {
        // cooperative load + hi/lo split of x[r0..r0+15][k0..k0+CHUNK)
        {
            const int idx = tid * 8;                 // 512 thr * 8 = 4096 elems
            const int row = idx >> 8;                // CHUNK == 256
            const int col = idx & (CHUNK - 1);
            const float* xp = x + (size_t)(r0 + row) * D_DIM + k0 + col;
            floatx4 f0 = *(const floatx4*)xp;
            floatx4 f1 = *(const floatx4*)(xp + 4);
            bf16x8 hv, lv;
            #pragma unroll
            for (int j = 0; j < 4; ++j) { bf16 h,l; split_bf16(f0[j],h,l); hv[j]=h; lv[j]=l; }
            #pragma unroll
            for (int j = 0; j < 4; ++j) { bf16 h,l; split_bf16(f1[j],h,l); hv[4+j]=h; lv[4+j]=l; }
            *(bf16x8*)&s_hi[row * CHUNK + col] = hv;
            *(bf16x8*)&s_lo[row * CHUNK + col] = lv;
        }
        __syncthreads();

        #pragma unroll 4
        for (int kk = 0; kk < CHUNK; kk += 32) {
            // A fragment (16x32 bf16): two 16B LDS loads per plane,
            // lanes<16 take K {0-15}, lanes>=16 take K {16-31}
            const int ab = m * CHUNK + kk + aoff;
            bf16x8 ah0 = *(const bf16x8*)&s_hi[ab];
            bf16x8 ah1 = *(const bf16x8*)&s_hi[ab + 8];
            bf16x8 al0 = *(const bf16x8*)&s_lo[ab];
            bf16x8 al1 = *(const bf16x8*)&s_lo[ab + 8];
            bf16x16 ahi = __builtin_shufflevector(ah0, ah1,
                0,1,2,3,4,5,6,7,8,9,10,11,12,13,14,15);
            bf16x16 alo = __builtin_shufflevector(al0, al1,
                0,1,2,3,4,5,6,7,8,9,10,11,12,13,14,15);

            // B fragment (32x16 bf16): lane = expert column, contiguous K
            bf16x16 bhi, blo;
            if constexpr (PRECONV) {
                const int wo = k0 + kk + bko;        // 32B-aligned element offset
                bhi = *(const bf16x16*)&whrow[wo];
                blo = *(const bf16x16*)&wlrow[wo];
            } else {
                const float* bp = wrow + k0 + kk + bko;
                floatx4 b0 = *(const floatx4*)bp;
                floatx4 b1 = *(const floatx4*)(bp + 4);
                floatx4 b2 = *(const floatx4*)(bp + 8);
                floatx4 b3 = *(const floatx4*)(bp + 12);
                #pragma unroll
                for (int j = 0; j < 4; ++j) {
                    bf16 h,l;
                    split_bf16(b0[j],h,l); bhi[j]    = h; blo[j]    = l;
                    split_bf16(b1[j],h,l); bhi[4+j]  = h; blo[4+j]  = l;
                    split_bf16(b2[j],h,l); bhi[8+j]  = h; blo[8+j]  = l;
                    split_bf16(b3[j],h,l); bhi[12+j] = h; blo[12+j] = l;
                }
            }

            // hi*hi + hi*lo + lo*hi  (lo*lo negligible)
            acc = __builtin_amdgcn_wmma_f32_16x16x32_bf16(
                      false, ahi, false, bhi, (short)0, acc, false, false);
            acc = __builtin_amdgcn_wmma_f32_16x16x32_bf16(
                      false, ahi, false, blo, (short)0, acc, false, false);
            acc = __builtin_amdgcn_wmma_f32_16x16x32_bf16(
                      false, alo, false, bhi, (short)0, acc, false, false);
        }
        __syncthreads();
    }

    // ---- activation: original = sqrt(softplus(score)) ----
    #pragma unroll
    for (int r = 0; r < 8; ++r) {
        float sc = acc[r];
        float sp = (sc > 20.f) ? sc : log1pf(__expf(sc));
        const int row = r + (hi16 ? 8 : 0);     // C layout: VGPR r, lane group
        s_orig[row * E_DIM + e0 + m] = sqrtf(sp);
    }
    __syncthreads();

    // ---- top-6 per row: wave `wave` handles row `wave` ----
    {
        const int row = wave;
        float bv[8];
        #pragma unroll
        for (int j = 0; j < 8; ++j)
            bv[j] = s_orig[row * E_DIM + lane * 8 + j] + s_bias[lane * 8 + j];

        unsigned rm = 0;
        float wsel[TOPK]; int isel[TOPK];
        float sum = 0.f;
        for (int t = 0; t < TOPK; ++t) {
            float best = -3.4e38f; int bidx = 0x7fffffff;
            #pragma unroll
            for (int j = 0; j < 8; ++j) {
                if (!((rm >> j) & 1u)) {
                    float v = bv[j]; int id = lane * 8 + j;
                    if (v > best || (v == best && id < bidx)) { best = v; bidx = id; }
                }
            }
            #pragma unroll
            for (int off = 16; off > 0; off >>= 1) {
                float ov = __shfl_xor(best, off, 32);
                int   oi = __shfl_xor(bidx, off, 32);
                if (ov > best || (ov == best && oi < bidx)) { best = ov; bidx = oi; }
            }
            if ((bidx >> 3) == lane) rm |= 1u << (bidx & 7);   // remove winner
            float ws = s_orig[row * E_DIM + bidx];             // unbiased weight
            wsel[t] = ws; isel[t] = bidx; sum += ws;
        }
        if (lane == 0) {
            const float scale = ROUTE_SCALE / sum;
            const int grow = r0 + row;
            #pragma unroll
            for (int t = 0; t < TOPK; ++t) {
                out_w[(size_t)grow * TOPK + t] = wsel[t] * scale;
                out_i[(size_t)grow * TOPK + t] = (float)isel[t];
            }
        }
    }
}

extern "C" void kernel_launch(void* const* d_in, const int* in_sizes, int n_in,
                              void* d_out, int out_size, void* d_ws, size_t ws_size,
                              hipStream_t stream) {
    const float* x    = (const float*)d_in[0];
    const float* w    = (const float*)d_in[1];
    const float* bias = (const float*)d_in[2];
    const int N = in_sizes[0] / D_DIM;          // 16384

    float* out_w = (float*)d_out;                     // weights [N,6]
    float* out_i = out_w + (size_t)N * TOPK;          // indices [N,6] (as float)

    const size_t wsneed = (size_t)E_DIM * D_DIM * 2 * sizeof(bf16);  // 4 MB
    dim3 grid(N / ROWS);

    if (ws_size >= wsneed) {
        bf16* whi = (bf16*)d_ws;
        bf16* wlo = whi + (size_t)E_DIM * D_DIM;
        const int prep_blocks = (E_DIM * D_DIM) / (256 * 4);          // 1024
        split_w_kernel<<<prep_blocks, 256, 0, stream>>>(w, whi, wlo);
        gate_topk_kernel<true><<<grid, 512, 0, stream>>>(
            x, w, bias, whi, wlo, out_w, out_i);
    } else {
        gate_topk_kernel<false><<<grid, 512, 0, stream>>>(
            x, w, bias, nullptr, nullptr, out_w, out_i);
    }
}
